// T5Attention_13099650253022
// MI455X (gfx1250) — matrix operand order repaired and checked by reading, not compiled, not run on hardware
//
#include <hip/hip_runtime.h>
#include <hip/hip_bf16.h>
#include <stdint.h>

typedef __attribute__((ext_vector_type(16))) __bf16 v16bf;
typedef __attribute__((ext_vector_type(8)))  float  v8f;
typedef __attribute__((ext_vector_type(4)))  int    v4i;

#define S_LEN 2048
#define HID   1024
#define NHEAD 16
#define HDIM  64
#define BATCH 2
#define LUTN  4095   // 2*S_LEN - 1 distinct relative positions

#define AS1 __attribute__((address_space(1)))
#define AS3 __attribute__((address_space(3)))

#if defined(__AMDGCN__) && __has_builtin(__builtin_amdgcn_global_load_async_to_lds_b128)
#define HAVE_ASYNC_LDS 1
#else
#define HAVE_ASYNC_LDS 0
#endif

__device__ __forceinline__ unsigned short f2bf(float f) {
  unsigned u = __float_as_uint(f);
  u += 0x7FFFu + ((u >> 16) & 1u);     // round-to-nearest-even to bf16
  return (unsigned short)(u >> 16);
}

union FragU {            // two 16B chunks <-> one 16x bf16 WMMA fragment
  uint4 q[2];
  v16bf v;
};

// 16-byte global->LDS copy: async (ASYNCcnt) when the toolchain exposes it.
__device__ __forceinline__ void copy16_g2l(const unsigned short* g,
                                           unsigned short* l) {
#if HAVE_ASYNC_LDS
  __builtin_amdgcn_global_load_async_to_lds_b128((AS1 v4i*)g, (AS3 v4i*)l,
                                                 0, 0);
#else
  *(uint4*)l = *(const uint4*)g;
#endif
}

__device__ __forceinline__ void async_wait0() {
#if HAVE_ASYNC_LDS
#if __has_builtin(__builtin_amdgcn_s_wait_asynccnt)
  __builtin_amdgcn_s_wait_asynccnt(0);
#else
  asm volatile("s_wait_asynccnt 0" ::: "memory");
#endif
#endif
}

// ---- 16-lane (DPP-row) all-reduce via row_ror rotations: stays on the VALU
// (co-executes with the WMMA pipe) instead of ds_bpermute through LDS.
#define DPP_ROR_F32(x, ctrl)                                              \
  __int_as_float(__builtin_amdgcn_update_dpp(0, __float_as_int(x), ctrl,  \
                                             0xF, 0xF, false))

__device__ __forceinline__ float rowmax16(float x) {
  x = fmaxf(x, DPP_ROR_F32(x, 0x128));   // row_ror:8
  x = fmaxf(x, DPP_ROR_F32(x, 0x124));   // row_ror:4
  x = fmaxf(x, DPP_ROR_F32(x, 0x122));   // row_ror:2
  x = fmaxf(x, DPP_ROR_F32(x, 0x121));   // row_ror:1
  return x;
}
__device__ __forceinline__ float rowsum16(float x) {
  x += DPP_ROR_F32(x, 0x128);
  x += DPP_ROR_F32(x, 0x124);
  x += DPP_ROR_F32(x, 0x122);
  x += DPP_ROR_F32(x, 0x121);
  return x;
}

// ---------------------------------------------------------------- fp32->bf16
__global__ void cvt_bf16_kernel(const float* __restrict__ src,
                                unsigned short* __restrict__ dst, int n) {
  for (int i = blockIdx.x * blockDim.x + threadIdx.x; i < n;
       i += gridDim.x * blockDim.x)
    dst[i] = f2bf(src[i]);
}

// ------------------------------------------------- relative-position bias LUT
// lut[h][idx], idx = (k - q) + (S_LEN-1); T5 bidirectional bucket formula.
__global__ void bias_lut_kernel(const float* __restrict__ rel,   // [32][16]
                                float* __restrict__ lut) {
  int idx = blockIdx.x * blockDim.x + threadIdx.x;
  if (idx >= LUTN) return;
  int d  = idx - (S_LEN - 1);
  int rb = (d > 0) ? 16 : 0;
  int ad = d < 0 ? -d : d;
  int bucket;
  if (ad < 8) {
    bucket = rb + ad;
  } else {
    float t  = __logf((float)ad * 0.125f) * (8.0f / __logf(16.0f));
    int   rl = 8 + (int)t;
    rl = rl > 15 ? 15 : rl;
    bucket = rb + rl;
  }
  for (int h = 0; h < NHEAD; ++h)
    lut[h * LUTN + idx] = rel[bucket * NHEAD + h];
}

// ------------------------------------------- materialize position_bias output
// outb[h][q][k] = lut[h][k - q + S-1]; float4 stores, bandwidth-bound.
__global__ void bias_out_kernel(const float* __restrict__ lut,
                                float* __restrict__ outb) {
  const int h = blockIdx.z, q = blockIdx.y;
  const int k = (blockIdx.x * blockDim.x + threadIdx.x) * 4;
  const float* lr = lut + (size_t)h * LUTN + (k - q + (S_LEN - 1));
  float4 v = make_float4(lr[0], lr[1], lr[2], lr[3]);
  *(float4*)(outb + ((size_t)h * S_LEN + q) * S_LEN + k) = v;
}

// --------------------------------------------------------- generic bf16 GEMM
// C[M=grid.y*64][N=grid.x*64] = A[M][1024] * B[1024][N], bf16 in, f32 acc.
// 128 threads = 4 waves; wave w owns rows w*16..w*16+15; 4 N-tiles per wave.
template <bool OUT_BF16>
__global__ void __launch_bounds__(128)
gemm_bf16_kernel(const unsigned short* __restrict__ A,
                 const unsigned short* __restrict__ Bm,
                 void* __restrict__ Cout) {
  __shared__ unsigned short ldsA[64][32];    // [m][k]
  __shared__ unsigned short ldsBt[64][32];   // [n][k]  (transposed)
  const int tid = threadIdx.x, wave = tid >> 5, lane = tid & 31;
  const int half = lane >> 4, l16 = lane & 15, ko = half * 8;
  const int m0 = blockIdx.y * 64, n0 = blockIdx.x * 64;

  v8f acc[4] = {};

  for (int k0 = 0; k0 < HID; k0 += 32) {
    __syncthreads();
    {   // A tile: 64x32 halves, 32B per thread (async DMA to LDS if available)
      int r = tid >> 1, c = (tid & 1) * 16;
      const unsigned short* gs = A + (size_t)(m0 + r) * HID + k0 + c;
      copy16_g2l(gs, &ldsA[r][c]);
      copy16_g2l(gs + 8, &ldsA[r][c + 8]);
    }
    {   // B tile transposed: ldsBt[n][k] = B[k0+k][n0+n]
      int k = tid >> 2, nb = (tid & 3) * 16;
      const unsigned short* gp = Bm + (size_t)(k0 + k) * HID + n0 + nb;
#pragma unroll
      for (int i = 0; i < 16; ++i) ldsBt[nb + i][k] = gp[i];
    }
    async_wait0();
    __syncthreads();

    FragU ua;                                   // A frag: row m0+wave*16+l16
    const int ar = wave * 16 + l16;
    ua.q[0] = *(const uint4*)&ldsA[ar][ko];
    ua.q[1] = *(const uint4*)&ldsA[ar][ko + 16];
#pragma unroll
    for (int nt = 0; nt < 4; ++nt) {
      FragU ub;                                 // B frag: col nt*16+l16
      const int br = nt * 16 + l16, boff = half * 8;
      ub.q[0] = *(const uint4*)&ldsBt[br][boff];
      ub.q[1] = *(const uint4*)&ldsBt[br][boff + 16];
      acc[nt] = __builtin_amdgcn_wmma_f32_16x16x32_bf16(
          false, ua.v, false, ub.v, (short)0, acc[nt], false, false);
    }
  }

#pragma unroll
  for (int nt = 0; nt < 4; ++nt)
#pragma unroll
    for (int e = 0; e < 8; ++e) {
      const size_t row = m0 + wave * 16 + half * 8 + e;
      const size_t col = n0 + nt * 16 + l16;
      if (OUT_BF16)
        ((unsigned short*)Cout)[row * HID + col] = f2bf(acc[nt][e]);
      else
        ((float*)Cout)[row * HID + col] = acc[nt][e];
    }
}

// -------------------------------------------------- flash attention (per b,h)
// grid: (S/64, NHEAD, BATCH); 128 thr = 4 waves; wave owns 16 q-rows.
__global__ void __launch_bounds__(128)
attn_flash_kernel(const unsigned short* __restrict__ Q,
                  const unsigned short* __restrict__ K,
                  const unsigned short* __restrict__ V,
                  const float* __restrict__ lut,
                  unsigned short* __restrict__ attnOut) {
  __shared__ float sBias[LUTN];                  // this head's bias row
  __shared__ unsigned short sK[32][64];          // [k][d]
  __shared__ unsigned short sVt[64][32];         // [d][k]
  __shared__ unsigned short sP[4][16][32];       // per-wave P (C->A relayout)

  const int tid = threadIdx.x, wave = tid >> 5, lane = tid & 31;
  const int half = lane >> 4, l16 = lane & 15, ko = half * 8;
  const int h = blockIdx.y, b = blockIdx.z;
  const int q0 = blockIdx.x * 64 + wave * 16;

  const unsigned short* Qb = Q + (size_t)b * S_LEN * HID + h * HDIM;
  const unsigned short* Kb = K + (size_t)b * S_LEN * HID + h * HDIM;
  const unsigned short* Vb = V + (size_t)b * S_LEN * HID + h * HDIM;

  for (int i = tid; i < LUTN; i += 128) sBias[i] = lut[(size_t)h * LUTN + i];

  FragU uq[2];                       // Q fragments, d = 0..31 and 32..63
  {
    const unsigned short* qr = Qb + (size_t)(q0 + l16) * HID;
    uq[0].q[0] = *(const uint4*)(qr + ko);
    uq[0].q[1] = *(const uint4*)(qr + ko + 16);
    uq[1].q[0] = *(const uint4*)(qr + 32 + ko);
    uq[1].q[1] = *(const uint4*)(qr + 32 + ko + 16);
  }

  float mrow[8], lrow[8];
#pragma unroll
  for (int e = 0; e < 8; ++e) { mrow[e] = -3.0e38f; lrow[e] = 0.0f; }
  v8f acc[4] = {};

  const int sr = tid >> 2, sc = (tid & 3) * 16;  // staging row / col chunk

  for (int k0 = 0; k0 < S_LEN; k0 += 32) {
    __syncthreads();
    {   // stage K tile (row-major) -- async DMA to LDS when available
      const unsigned short* gs = Kb + (size_t)(k0 + sr) * HID + sc;
      copy16_g2l(gs, &sK[sr][sc]);
      copy16_g2l(gs + 8, &sK[sr][sc + 8]);
    }
    {   // stage V tile transposed: sVt[d][k]
      const unsigned short* gp = Vb + (size_t)(k0 + sr) * HID + sc;
#pragma unroll
      for (int i = 0; i < 16; ++i) sVt[sc + i][sr] = gp[i];
    }
    if (k0 + 32 < S_LEN) {   // prefetch next tiles (global_prefetch_b8)
      __builtin_prefetch(Kb + (size_t)(k0 + 32 + sr) * HID + sc, 0, 1);
      __builtin_prefetch(Vb + (size_t)(k0 + 32 + sr) * HID + sc, 0, 1);
    }
    async_wait0();
    __syncthreads();

    // scores S[16q][32k] = Q(16x64) * K^T : 2 kcol-tiles x 2 d-steps
    v8f s[2] = {};
#pragma unroll
    for (int kt = 0; kt < 2; ++kt) {
      const int br = kt * 16 + l16;
#pragma unroll
      for (int ds = 0; ds < 2; ++ds) {
        FragU ub;
        const int boff = ds * 32 + half * 8;
        ub.q[0] = *(const uint4*)&sK[br][boff];
        ub.q[1] = *(const uint4*)&sK[br][boff + 16];
        s[kt] = __builtin_amdgcn_wmma_f32_16x16x32_bf16(
            false, uq[ds].v, false, ub.v, (short)0, s[kt], false, false);
      }
    }

    // bias add + online softmax (DPP row_ror reductions across 16-lane rows)
#pragma unroll
    for (int e = 0; e < 8; ++e) {
      const int qrow = q0 + half * 8 + e;
      float s0 = s[0][e] + sBias[k0 + l16      - qrow + (S_LEN - 1)];
      float s1 = s[1][e] + sBias[k0 + 16 + l16 - qrow + (S_LEN - 1)];
      const float mx = rowmax16(fmaxf(s0, s1));
      const float mn = fmaxf(mrow[e], mx);
      const float p0 = __expf(s0 - mn);
      const float p1 = __expf(s1 - mn);
      const float sc2 = __expf(mrow[e] - mn);
      const float rs = rowsum16(p0 + p1);
      lrow[e] = lrow[e] * sc2 + rs;
      mrow[e] = mn;
      sP[wave][half * 8 + e][l16]      = f2bf(p0);
      sP[wave][half * 8 + e][16 + l16] = f2bf(p1);
#pragma unroll
      for (int nt = 0; nt < 4; ++nt) acc[nt][e] *= sc2;
    }
    __syncthreads();   // make per-wave P LDS visible cross-lane

    // acc += P(16x32) * V(32x64)
    FragU up;
    up.q[0] = *(const uint4*)&sP[wave][l16][ko];
    up.q[1] = *(const uint4*)&sP[wave][l16][ko + 16];
#pragma unroll
    for (int nt = 0; nt < 4; ++nt) {
      FragU ub;
      const int br = nt * 16 + l16, boff = half * 8;
      ub.q[0] = *(const uint4*)&sVt[br][boff];
      ub.q[1] = *(const uint4*)&sVt[br][boff + 16];
      acc[nt] = __builtin_amdgcn_wmma_f32_16x16x32_bf16(
          false, up.v, false, ub.v, (short)0, acc[nt], false, false);
    }
  }

  // finalize: divide by l, merge heads, store bf16 for the O-projection GEMM
#pragma unroll
  for (int e = 0; e < 8; ++e) {
    const float inv = 1.0f / lrow[e];
    const size_t row = q0 + half * 8 + e;
#pragma unroll
    for (int nt = 0; nt < 4; ++nt) {
      const size_t col = (size_t)h * HDIM + nt * 16 + l16;
      attnOut[((size_t)b * S_LEN + row) * HID + col] = f2bf(acc[nt][e] * inv);
    }
  }
}

// ---------------------------------------------------------------------------
extern "C" void kernel_launch(void* const* d_in, const int* in_sizes, int n_in,
                              void* d_out, int out_size, void* d_ws,
                              size_t ws_size, hipStream_t stream) {
  (void)in_sizes; (void)n_in; (void)out_size; (void)ws_size;
  const float* hs  = (const float*)d_in[0];
  const float* Wq  = (const float*)d_in[1];
  const float* Wk  = (const float*)d_in[2];
  const float* Wv  = (const float*)d_in[3];
  const float* Wo  = (const float*)d_in[4];
  const float* rel = (const float*)d_in[5];

  float* out     = (float*)d_out;                       // [2,2048,1024]
  float* outBias = out + (size_t)BATCH * S_LEN * HID;   // [1,16,2048,2048]

  char* ws = (char*)d_ws;
  const size_t MB = 1ull << 20;
  unsigned short* hsb = (unsigned short*)(ws);            //  8 MB bf16 hidden
  unsigned short* wqb = (unsigned short*)(ws +  8 * MB);  //  2 MB
  unsigned short* wkb = (unsigned short*)(ws + 10 * MB);
  unsigned short* wvb = (unsigned short*)(ws + 12 * MB);
  unsigned short* wob = (unsigned short*)(ws + 14 * MB);
  unsigned short* qb  = (unsigned short*)(ws + 16 * MB);  //  8 MB
  unsigned short* kb  = (unsigned short*)(ws + 24 * MB);
  unsigned short* vb  = (unsigned short*)(ws + 32 * MB);
  unsigned short* ab  = (unsigned short*)(ws + 40 * MB);
  float*          lut = (float*)(ws + 48 * MB);           // 16*4095 floats

  const int NHS = BATCH * S_LEN * HID;
  cvt_bf16_kernel<<<1024, 256, 0, stream>>>(hs, hsb, NHS);
  cvt_bf16_kernel<<<512, 256, 0, stream>>>(Wq, wqb, HID * HID);
  cvt_bf16_kernel<<<512, 256, 0, stream>>>(Wk, wkb, HID * HID);
  cvt_bf16_kernel<<<512, 256, 0, stream>>>(Wv, wvb, HID * HID);
  cvt_bf16_kernel<<<512, 256, 0, stream>>>(Wo, wob, HID * HID);

  bias_lut_kernel<<<(LUTN + 127) / 128, 128, 0, stream>>>(rel, lut);
  bias_out_kernel<<<dim3(S_LEN / (256 * 4), S_LEN, NHEAD), 256, 0, stream>>>(
      lut, outBias);

  dim3 gg(HID / 64, (BATCH * S_LEN) / 64);   // (16, 64) blocks of 128 thr
  gemm_bf16_kernel<true><<<gg, 128, 0, stream>>>(hsb, wqb, qb);
  gemm_bf16_kernel<true><<<gg, 128, 0, stream>>>(hsb, wkb, kb);
  gemm_bf16_kernel<true><<<gg, 128, 0, stream>>>(hsb, wvb, vb);

  attn_flash_kernel<<<dim3(S_LEN / 64, NHEAD, BATCH), 128, 0, stream>>>(
      qb, kb, vb, lut, ab);

  gemm_bf16_kernel<false><<<gg, 128, 0, stream>>>(ab, wob, out);
}
